// MultiHeadAttention_5875515261590
// MI455X (gfx1250) — matrix operand order repaired and checked by reading, not compiled, not run on hardware
//
#include <hip/hip_runtime.h>
#include <stdint.h>

typedef __bf16 bf16_t;
typedef bf16_t v16bf __attribute__((ext_vector_type(16)));
typedef bf16_t v8bf  __attribute__((ext_vector_type(8)));
typedef bf16_t v2bf  __attribute__((ext_vector_type(2)));
typedef float  v8f   __attribute__((ext_vector_type(8)));

#define DEV static __device__ __forceinline__

DEV v8f wmma_bf16(v16bf a, v16bf b, v8f c) {
  // D = A(16x32 bf16) * B(32x16 bf16) + C(16x16 f32)
  return __builtin_amdgcn_wmma_f32_16x16x32_bf16(false, a, false, b, (short)0, c,
                                                 false, false);
}

DEV v8f zero8() { v8f z = {0.f,0.f,0.f,0.f,0.f,0.f,0.f,0.f}; return z; }

// raw v_exp_f32: args are always <= 0 in softmax; underflow flushes to 0 (wanted)
DEV float fast_exp2(float x) { return __builtin_amdgcn_exp2f(x); }

union Frag { v16bf v; v8bf h[2]; unsigned u[8]; };

// ---- A fragment (16x32): lane = M row (lane&15); elements 0..15 hold
// K=kg..kg+15, kg = (lane>=16 ? 16 : 0). Two 16B loads.
// Works for global or LDS source (addrspace inferred after inlining).
DEV v16bf load_frag_a(const bf16_t* base, int ld) {
  int lane = threadIdx.x & 31;
  int m  = lane & 15;
  int kg = (lane >> 4) << 4;
  const bf16_t* p = base + (size_t)m * ld;
  Frag f;
  f.h[0] = *(const v8bf*)(p + kg);
  f.h[1] = *(const v8bf*)(p + kg + 8);
  return f.v;
}

// ---- B fragment (32x16): B[k][n] = rowmat[n][k], rowmat row-major stride ld
// ("NT" layout). lane = N col; lanes 0-15 hold K=0..15, lanes 16-31 K=16..31.
DEV v16bf load_frag_b(const bf16_t* base, int ld) {
  int lane = threadIdx.x & 31;
  int n  = lane & 15;
  int kb = (lane >> 4) << 4;
  const bf16_t* p = base + (size_t)n * ld + kb;
  Frag f;
  f.h[0] = *(const v8bf*)(p);
  f.h[1] = *(const v8bf*)(p + 8);
  return f.v;
}

DEV unsigned pack_bf16x2(float lo, float hi) {
  union { v2bf v; unsigned u; } t;
  t.v = v2bf{(bf16_t)lo, (bf16_t)hi};
  return t.u;
}

// Async 16B global -> LDS copy (CDNA5 async path, tracked by ASYNCcnt).
// LDS offset = low 32 bits of the generic pointer (aperture layout).
DEV void async_copy_b128(const bf16_t* gsrc, bf16_t* ldst) {
  unsigned long long ga = (unsigned long long)(uintptr_t)gsrc;
  unsigned la = (unsigned)(uintptr_t)ldst;
  asm volatile("global_load_async_to_lds_b128 %0, %1, off"
               :: "v"(la), "v"(ga) : "memory");
}

// ============================================================================
// fp32 -> bf16 cast, 8 elements per thread.
// ============================================================================
__global__ void __launch_bounds__(256) cast_bf16_kernel(const float* __restrict__ src,
                                                        bf16_t* __restrict__ dst,
                                                        int n8) {
  int i = blockIdx.x * blockDim.x + threadIdx.x;
  if (i >= n8) return;
  const float* p = src + (size_t)i * 8;
  v8bf o;
#pragma unroll
  for (int j = 0; j < 8; ++j) o[j] = (bf16_t)p[j];
  *(v8bf*)(dst + (size_t)i * 8) = o;
}

// ============================================================================
// Projection GEMM: Y[M,N] = X[M,K] @ W[N,K]^T + bias, per-wave 32x64 tile,
// double-buffered fragment loads. X and W are pre-cast bf16.
// MODE 0: out bf16 scattered to [B,H,S,64]   (Q, K heads)
// MODE 1: out bf16 scattered to [B,H,64,S]   (V heads, transposed), packed stores
// MODE 2: out f32 dense [M,N]                (final output projection)
// ============================================================================
template<int MODE>
__global__ void __launch_bounds__(256) proj_kernel(const bf16_t* __restrict__ X,
                                                   const bf16_t* __restrict__ W,
                                                   const float* __restrict__ bias,
                                                   void* __restrict__ outp) {
  const int N = 1024, K = 1024;
  int wid = (blockIdx.x * blockDim.x + threadIdx.x) >> 5;
  int tn = (wid & 15) << 6;   // 16 N-tiles of 64
  int tm = (wid >> 4) << 5;   // 128 M-tiles of 32
  int lane = threadIdx.x & 31;

  v8f acc[2][4];
#pragma unroll
  for (int i = 0; i < 2; ++i)
#pragma unroll
    for (int j = 0; j < 4; ++j) acc[i][j] = zero8();

  // software pipeline: preload k0 = 0
  v16bf na0 = load_frag_a(X + (size_t)tm * K, K);
  v16bf na1 = load_frag_a(X + (size_t)(tm + 16) * K, K);
  v16bf nb0 = load_frag_b(W + (size_t)(tn +  0) * K, K);
  v16bf nb1 = load_frag_b(W + (size_t)(tn + 16) * K, K);
  v16bf nb2 = load_frag_b(W + (size_t)(tn + 32) * K, K);
  v16bf nb3 = load_frag_b(W + (size_t)(tn + 48) * K, K);

  for (int k0 = 0; k0 < K; k0 += 32) {
    v16bf a0 = na0, a1 = na1;
    v16bf b0 = nb0, b1 = nb1, b2 = nb2, b3 = nb3;
    if (k0 + 32 < K) {
      int kn = k0 + 32;
      na0 = load_frag_a(X + (size_t)tm * K + kn, K);
      na1 = load_frag_a(X + (size_t)(tm + 16) * K + kn, K);
      nb0 = load_frag_b(W + (size_t)(tn +  0) * K + kn, K);
      nb1 = load_frag_b(W + (size_t)(tn + 16) * K + kn, K);
      nb2 = load_frag_b(W + (size_t)(tn + 32) * K + kn, K);
      nb3 = load_frag_b(W + (size_t)(tn + 48) * K + kn, K);
    }
    acc[0][0] = wmma_bf16(a0, b0, acc[0][0]);
    acc[1][0] = wmma_bf16(a1, b0, acc[1][0]);
    acc[0][1] = wmma_bf16(a0, b1, acc[0][1]);
    acc[1][1] = wmma_bf16(a1, b1, acc[1][1]);
    acc[0][2] = wmma_bf16(a0, b2, acc[0][2]);
    acc[1][2] = wmma_bf16(a1, b2, acc[1][2]);
    acc[0][3] = wmma_bf16(a0, b3, acc[0][3]);
    acc[1][3] = wmma_bf16(a1, b3, acc[1][3]);
  }

  // C/D layout: VGPR r -> row r (lanes 0-15) or 8+r (lanes 16-31); col = lane&15
  int n16 = lane & 15;
  int mh  = (lane >> 4) << 3;
#pragma unroll
  for (int mi = 0; mi < 2; ++mi)
#pragma unroll
    for (int j = 0; j < 4; ++j) {
      int col = tn + j * 16 + n16;
      float bv = bias[col];
      if (MODE == 1) {
        // rows tm+mi*16+mh+r are contiguous in s -> pack one 16B store
        int row0 = tm + mi * 16 + mh;
        int bb = row0 >> 11, s0 = row0 & 2047;
        int h = col >> 6, dk = col & 63;
        v8bf pk;
#pragma unroll
        for (int r = 0; r < 8; ++r) pk[r] = (bf16_t)(acc[mi][j][r] + bv);
        bf16_t* o = (bf16_t*)outp;
        *(v8bf*)(o + (((size_t)(bb * 16 + h)) * 64 + dk) * 2048 + s0) = pk;
      } else {
#pragma unroll
        for (int r = 0; r < 8; ++r) {
          int row = tm + mi * 16 + mh + r;
          float val = acc[mi][j][r] + bv;
          if (MODE == 2) {
            ((float*)outp)[(size_t)row * N + col] = val;
          } else {
            int bb = row >> 11, s = row & 2047;
            int h  = col >> 6,  dk = col & 63;
            ((bf16_t*)outp)[(((size_t)(bb * 16 + h)) * 2048 + s) * 64 + dk] =
                (bf16_t)val;
          }
        }
      }
    }
}

// ============================================================================
// Block-cooperative transposed flash attention.
// Block = 8 waves, all on the same (b,h); wave i owns q-rows [row0, row0+16).
// K/V tiles (64 keys) staged into double-buffered LDS with async global->LDS
// loads (ASYNCcnt-pipelined), then each wave reads WMMA fragments from LDS.
// Scores computed transposed (S^T = K_blk @ Q^T): lane owns ONE q-row, keys on
// VGPRs -> softmax is in-lane VALU + one xor-16 shuffle; running stats scalar.
// P^T assembled in registers; O accumulated transposed: O^T = V^T @ P^T.
// Qh,Kh: [B*H,S,64] bf16.  Vt: [B*H,64,S] bf16.  Xo: [B,S,1024] bf16.
// ============================================================================
__global__ void __launch_bounds__(256) attn_kernel(const bf16_t* __restrict__ Qh,
                                                   const bf16_t* __restrict__ Kh,
                                                   const bf16_t* __restrict__ Vt,
                                                   bf16_t* __restrict__ Xo) {
  const int S  = 2048;
  const int KB = 64;                                  // keys per staged tile
  const float SC = 0.125f * 1.44269504088896340736f;  // 1/sqrt(64) * log2(e)

  __shared__ __align__(16) bf16_t sK[2][KB * 64];     // [key][d]   8KB each
  __shared__ __align__(16) bf16_t sV[2][64 * KB];     // [d][key]   8KB each

  int tid  = threadIdx.x;
  int wib  = tid >> 5;
  int lane = tid & 31;
  int qn   = lane & 15;     // q-row owned by this lane
  int half = lane >> 4;     // key half-group

  int bh   = blockIdx.x >> 4;   // 16 blocks per (b,h)
  int qblk = blockIdx.x & 15;
  int row0 = (qblk * 8 + wib) * 16;

  const bf16_t* Q = Qh + (size_t)bh * S * 64 + (size_t)row0 * 64;
  const bf16_t* K = Kh + (size_t)bh * S * 64;
  const bf16_t* V = Vt + (size_t)bh * 64 * S;

  v16bf bq0 = load_frag_b(Q, 64);       // B[k=d 0..31][n=qrow]
  v16bf bq1 = load_frag_b(Q + 32, 64);  // d 32..63

  v8f o0 = zero8(), o1 = zero8(), o2 = zero8(), o3 = zero8();
  float mprev = -1e30f, lsum = 0.f;

  // cooperative async stage of one 64-key K tile + V^T tile (4 x 16B / thread)
  auto stage = [&](int kb, int buf) {
#pragma unroll
    for (int t = 0; t < 2; ++t) {
      int s = tid + t * 256;                 // 512 16B slots
      int key = s >> 3, dq = (s & 7) << 3;
      async_copy_b128(K + (size_t)(kb + key) * 64 + dq, &sK[buf][key * 64 + dq]);
    }
#pragma unroll
    for (int t = 0; t < 2; ++t) {
      int s = tid + t * 256;
      int d = s >> 3, ko = (s & 7) << 3;
      async_copy_b128(V + (size_t)d * S + kb + ko, &sV[buf][d * KB + ko]);
    }
  };

  stage(0, 0);
  const int NBLK = S / KB;   // 32
  for (int i = 0; i < NBLK; ++i) {
    int cur = i & 1;
    if (i + 1 < NBLK) {
      stage((i + 1) * KB, cur ^ 1);
      // async loads retire in order: <=4 outstanding => tile i has landed
      asm volatile("s_wait_asynccnt 0x4" ::: "memory");
    } else {
      asm volatile("s_wait_asynccnt 0x0" ::: "memory");
    }
    __syncthreads();   // tile `cur` visible to all waves

#pragma unroll
    for (int sub = 0; sub < 2; ++sub) {      // two 32-key sub-blocks
      const bf16_t* Kt = &sK[cur][sub * 32 * 64];
      const bf16_t* Vw = &sV[cur][sub * 32];

      // S^T tiles: keys 0..15 (t0) and 16..31 (t1) of sub-block, qrows on lanes
      v16bf ak00 = load_frag_a(Kt, 64);
      v16bf ak01 = load_frag_a(Kt + 32, 64);
      v16bf ak10 = load_frag_a(Kt + 16 * 64, 64);
      v16bf ak11 = load_frag_a(Kt + 16 * 64 + 32, 64);
      v8f t0 = zero8();
      t0 = wmma_bf16(ak00, bq0, t0);
      t0 = wmma_bf16(ak01, bq1, t0);
      v8f t1 = zero8();
      t1 = wmma_bf16(ak10, bq0, t1);
      t1 = wmma_bf16(ak11, bq1, t1);

      // V^T fragments (issued early; overlap softmax VALU)
      v16bf av0 = load_frag_a(Vw, 64);
      v16bf av1 = load_frag_a(Vw + 16 * 64, 64);
      v16bf av2 = load_frag_a(Vw + 32 * 64, 64);
      v16bf av3 = load_frag_a(Vw + 48 * 64, 64);

      // ---- online softmax in log2 domain (lane = one q-row)
      float x[16];
#pragma unroll
      for (int i2 = 0; i2 < 8; ++i2) { x[i2] = t0[i2] * SC; x[8 + i2] = t1[i2] * SC; }
      float mx = x[0];
#pragma unroll
      for (int i2 = 1; i2 < 16; ++i2) mx = fmaxf(mx, x[i2]);
      mx = fmaxf(mx, __shfl_xor(mx, 16, 32));
      float mnew  = fmaxf(mprev, mx);
      float alpha = fast_exp2(mprev - mnew);
      mprev = mnew;
      float e[16], sum = 0.f;
#pragma unroll
      for (int i2 = 0; i2 < 16; ++i2) { e[i2] = fast_exp2(x[i2] - mnew); sum += e[i2]; }
      sum += __shfl_xor(sum, 16, 32);
      lsum = lsum * alpha + sum;
#pragma unroll
      for (int r = 0; r < 8; ++r) {
        o0[r] *= alpha; o1[r] *= alpha; o2[r] *= alpha; o3[r] *= alpha;
      }

      // ---- assemble P^T B-fragment in registers
      unsigned g0[4], g1[4], rA[4], rB[4];
#pragma unroll
      for (int d = 0; d < 4; ++d) {
        g0[d] = pack_bf16x2(e[2 * d], e[2 * d + 1]);
        g1[d] = pack_bf16x2(e[8 + 2 * d], e[8 + 2 * d + 1]);
      }
#pragma unroll
      for (int d = 0; d < 4; ++d) {
        rA[d] = __shfl_xor(g0[d], 16, 32);
        rB[d] = __shfl_xor(g1[d], 16, 32);
      }
      Frag bp;
      bool hi = (half != 0);
#pragma unroll
      for (int d = 0; d < 4; ++d) {
        bp.u[d]     = hi ? rB[d] : g0[d];
        bp.u[4 + d] = hi ? g1[d] : rA[d];
      }

      // ---- O^T += V^T @ P^T
      o0 = wmma_bf16(av0, bp.v, o0);
      o1 = wmma_bf16(av1, bp.v, o1);
      o2 = wmma_bf16(av2, bp.v, o2);
      o3 = wmma_bf16(av3, bp.v, o3);
    }
    __syncthreads();   // all waves done with tile `cur` before it is re-staged
  }

  // O^T: VGPR r = d offset (half*8 + r within 16-group), lane = qrow
  float linv = __builtin_amdgcn_rcpf(lsum);
  int b = bh >> 4, h = bh & 15;
  bf16_t* orow = Xo + ((size_t)(b * 2048 + row0 + qn)) * 1024 +
                 (size_t)h * 64 + (size_t)half * 8;
  v8f oo[4] = {o0, o1, o2, o3};
#pragma unroll
  for (int j = 0; j < 4; ++j) {
    v8bf pk;
#pragma unroll
    for (int r = 0; r < 8; ++r) pk[r] = (bf16_t)(oo[j][r] * linv);
    *(v8bf*)(orow + j * 16) = pk;
  }
}

extern "C" void kernel_launch(void* const* d_in, const int* in_sizes, int n_in,
                              void* d_out, int out_size, void* d_ws, size_t ws_size,
                              hipStream_t stream) {
  (void)in_sizes; (void)n_in; (void)out_size; (void)ws_size;
  const float* q  = (const float*)d_in[0];
  const float* k  = (const float*)d_in[1];
  const float* v  = (const float*)d_in[2];
  const float* Wq = (const float*)d_in[3];
  const float* bq = (const float*)d_in[4];
  const float* Wk = (const float*)d_in[5];
  const float* bk = (const float*)d_in[6];
  const float* Wv = (const float*)d_in[7];
  const float* bv = (const float*)d_in[8];
  const float* Wo = (const float*)d_in[9];
  const float* bo = (const float*)d_in[10];

  const size_t headBytes = (size_t)32 * 2048 * 64 * sizeof(bf16_t);   // 8 MB
  const size_t wBytes    = (size_t)1024 * 1024 * sizeof(bf16_t);      // 2 MB
  const size_t actBytes  = (size_t)4096 * 1024 * sizeof(bf16_t);      // 8 MB
  char* ws = (char*)d_ws;
  bf16_t* Qh  = (bf16_t*)(ws);
  bf16_t* Kh  = (bf16_t*)(ws + headBytes);
  bf16_t* Vt  = (bf16_t*)(ws + 2 * headBytes);
  bf16_t* Xa  = (bf16_t*)(ws + 3 * headBytes);
  bf16_t* Wqb = (bf16_t*)(ws + 4 * headBytes);
  bf16_t* Wkb = (bf16_t*)(ws + 4 * headBytes + wBytes);
  bf16_t* Wvb = (bf16_t*)(ws + 4 * headBytes + 2 * wBytes);
  bf16_t* Wob = (bf16_t*)(ws + 4 * headBytes + 3 * wBytes);
  bf16_t* qb  = (bf16_t*)(ws + 4 * headBytes + 4 * wBytes);
  bf16_t* kb  = (bf16_t*)(ws + 4 * headBytes + 4 * wBytes + actBytes);
  bf16_t* vb  = (bf16_t*)(ws + 4 * headBytes + 4 * wBytes + 2 * actBytes);

  dim3 blk(256);
  const int n8w = (1024 * 1024) / 8;
  const int n8a = (4096 * 1024) / 8;
  cast_bf16_kernel<<<dim3(n8w / 256), blk, 0, stream>>>(Wq, Wqb, n8w);
  cast_bf16_kernel<<<dim3(n8w / 256), blk, 0, stream>>>(Wk, Wkb, n8w);
  cast_bf16_kernel<<<dim3(n8w / 256), blk, 0, stream>>>(Wv, Wvb, n8w);
  cast_bf16_kernel<<<dim3(n8w / 256), blk, 0, stream>>>(Wo, Wob, n8w);
  cast_bf16_kernel<<<dim3(n8a / 256), blk, 0, stream>>>(q, qb, n8a);
  cast_bf16_kernel<<<dim3(n8a / 256), blk, 0, stream>>>(k, kb, n8a);
  cast_bf16_kernel<<<dim3(n8a / 256), blk, 0, stream>>>(v, vb, n8a);

  dim3 gproj(256);   // (4096/32)*(1024/64) = 2048 waves
  proj_kernel<0><<<gproj, blk, 0, stream>>>(qb, Wqb, bq, Qh);
  proj_kernel<0><<<gproj, blk, 0, stream>>>(kb, Wkb, bk, Kh);
  proj_kernel<1><<<gproj, blk, 0, stream>>>(vb, Wvb, bv, Vt);
  attn_kernel<<<512, blk, 0, stream>>>(Qh, Kh, Vt, Xa);   // 512 blocks x 8 waves
  proj_kernel<2><<<gproj, blk, 0, stream>>>(Xa, Wob, bo, d_out);
}
